// FastTransformer_12309376270776
// MI455X (gfx1250) — compile-verified, matrix operand order checked by reading, not run on hardware
//
#include <hip/hip_runtime.h>
#include <cstdint>

#define USE_ASYNC 1

// ---------------------------------------------------------------------------
// Types for CDNA5 WMMA (wave32, 16x16x32 bf16 -> f32)
// ---------------------------------------------------------------------------
typedef __attribute__((ext_vector_type(16))) __bf16 v16bf;
typedef __attribute__((ext_vector_type(8)))  float  v8f;

union FragU { uint4 q[2]; unsigned int u[8]; v16bf v; };

__device__ __forceinline__ unsigned short f2bf(float f) {
    unsigned int u = __float_as_uint(f);
    unsigned int r = u + 0x7FFFu + ((u >> 16) & 1u);   // round-to-nearest-even
    return (unsigned short)(r >> 16);
}

// ---------------------------------------------------------------------------
// bf16 NT GEMM:  C[M,N] = A[M,K] * B[N,K]^T (+ epilogue)
//   A, B are bf16 row-major with leading dims lda, ldb (halves).
//   EPI: 0 plain | 1 bias+BN | 2 bias+BN+relu | 3 bias+BN+relu+resCoef*res
//        4 bias+GELU(exact) | 5 bias
//   OUT: 0 f32 [M,ldc] | 1 bf16 [M,ldc] | 2 bf16 transposed [N-major, ldc]
// Block = 128 threads (4 waves), tile 64x64, K-step 32.
// Requires M%64==0, N%64==0, K%32==0, 16B-aligned A/B rows (lda%8==0).
// ---------------------------------------------------------------------------
template <int EPI, int OUT>
__global__ __launch_bounds__(128) void gemm_bf16_wmma(
    const unsigned short* __restrict__ A, const unsigned short* __restrict__ B,
    void* __restrict__ Co, int M, int N, int K,
    int lda, int ldb, int ldc,
    long long sA, long long sB, long long sC,
    const float* __restrict__ bias, const float* __restrict__ gamma,
    const float* __restrict__ beta, const float* __restrict__ res, float resCoef)
{
    __shared__ unsigned short As[64 * 40];   // 80B rows: 16B-aligned, conflict-free
    __shared__ unsigned short Bs[64 * 40];

    const int tid = threadIdx.x;
    const int m0  = blockIdx.y * 64;
    const int n0  = blockIdx.x * 64;
    const unsigned short* Ab = A + (long long)blockIdx.z * sA;
    const unsigned short* Bb = B + (long long)blockIdx.z * sB;

    const int lane = tid & 31;
    const int wave = tid >> 5;
    const int wm   = (wave >> 1) * 32;
    const int wn   = (wave & 1) * 32;
    const int half = lane >> 4;
    const int l16  = lane & 15;

    v8f acc[2][2];
#pragma unroll
    for (int i = 0; i < 2; ++i)
#pragma unroll
        for (int j = 0; j < 2; ++j)
#pragma unroll
            for (int r = 0; r < 8; ++r) acc[i][j][r] = 0.0f;

    for (int k0 = 0; k0 < K; k0 += 32) {
        // ---- stage 64x32-half tiles: pure 16B copies ----
#if USE_ASYNC
#pragma unroll
        for (int it = 0; it < 2; ++it) {
            int cid = it * 128 + tid;            // 256 chunks of 8 halves
            int row = cid >> 2, kb = (cid & 3) * 8;
            {
                unsigned lo = (unsigned)(unsigned long long)(const void*)&As[row * 40 + kb];
                unsigned long long ga =
                    (unsigned long long)(const void*)(Ab + (size_t)(m0 + row) * lda + k0 + kb);
                asm volatile("global_load_async_to_lds_b128 %0, %1, off"
                             :: "v"(lo), "v"(ga) : "memory");
            }
            {
                unsigned lo = (unsigned)(unsigned long long)(const void*)&Bs[row * 40 + kb];
                unsigned long long ga =
                    (unsigned long long)(const void*)(Bb + (size_t)(n0 + row) * ldb + k0 + kb);
                asm volatile("global_load_async_to_lds_b128 %0, %1, off"
                             :: "v"(lo), "v"(ga) : "memory");
            }
        }
        asm volatile("s_wait_asynccnt 0x0" ::: "memory");
#else
#pragma unroll
        for (int it = 0; it < 2; ++it) {
            int cid = it * 128 + tid;
            int row = cid >> 2, kb = (cid & 3) * 8;
            *(uint4*)&As[row * 40 + kb] =
                *(const uint4*)(Ab + (size_t)(m0 + row) * lda + k0 + kb);
            *(uint4*)&Bs[row * 40 + kb] =
                *(const uint4*)(Bb + (size_t)(n0 + row) * ldb + k0 + kb);
        }
#endif
        __syncthreads();

        // ---- fragments (ISA 16-bit layout: half0 K0-7 & 16-23, half1 K8-15 & 24-31) ----
        FragU af[2], bf[2];
#pragma unroll
        for (int i = 0; i < 2; ++i) {
            int row = wm + i * 16 + l16;
            af[i].q[0] = *(const uint4*)&As[row * 40 + half * 8];
            af[i].q[1] = *(const uint4*)&As[row * 40 + 16 + half * 8];
        }
#pragma unroll
        for (int j = 0; j < 2; ++j) {
            int col = wn + j * 16 + l16;
            bf[j].q[0] = *(const uint4*)&Bs[col * 40 + half * 8];
            bf[j].q[1] = *(const uint4*)&Bs[col * 40 + 16 + half * 8];
        }

#pragma unroll
        for (int i = 0; i < 2; ++i)
#pragma unroll
            for (int j = 0; j < 2; ++j)
                acc[i][j] = __builtin_amdgcn_wmma_f32_16x16x32_bf16(
                    false, af[i].v, false, bf[j].v, (short)0, acc[i][j], false, false);

        __syncthreads();
    }

    // ---- epilogue ----
    const float bnrs = rsqrtf(1.0f + 1e-5f);
#pragma unroll
    for (int j = 0; j < 2; ++j) {
        int gc = n0 + wn + j * 16 + l16;
        float bb = 0.f, sc = 1.f, bt = 0.f;
        if (EPI >= 1 && EPI <= 3) { bb = bias[gc]; sc = gamma[gc] * bnrs; bt = beta[gc]; }
        if (EPI >= 4)             { bb = bias[gc]; }
#pragma unroll
        for (int i = 0; i < 2; ++i) {
#pragma unroll
            for (int r = 0; r < 8; ++r) {
                int gr = m0 + wm + i * 16 + r + 8 * half;
                float v = acc[i][j][r];
                if (EPI == 1) v = (v + bb) * sc + bt;
                if (EPI == 2) v = fmaxf((v + bb) * sc + bt, 0.0f);
                if (EPI == 3) v = fmaxf((v + bb) * sc + bt, 0.0f)
                               + resCoef * res[(size_t)gr * N + gc];
                if (EPI == 4) { float z = v + bb; v = 0.5f * z * (1.0f + erff(z * 0.70710678118f)); }
                if (EPI == 5) v += bb;
                if (OUT == 0)
                    ((float*)Co)[(long long)blockIdx.z * sC + (size_t)gr * ldc + gc] = v;
                if (OUT == 1)
                    ((unsigned short*)Co)[(long long)blockIdx.z * sC + (size_t)gr * ldc + gc] = f2bf(v);
                if (OUT == 2)
                    ((unsigned short*)Co)[(long long)blockIdx.z * sC + (size_t)gc * ldc + gr] = f2bf(v);
            }
        }
    }
}

// ---------------------------------------------------------------------------
// Direct 3x3 stride-2 pad-1 conv. Input NCHW (flag) or NHWC; output NHWC f32.
// ---------------------------------------------------------------------------
__global__ void conv3x3_s2_k(const float* __restrict__ in, const float* __restrict__ w,
                             const float* __restrict__ bias, float* __restrict__ out,
                             int Cin, int Hin, int Cout, int Hout, int nchw,
                             long long nStrideIn)
{
    int gid = blockIdx.x * blockDim.x + threadIdx.x;
    int total = 4 * Hout * Hout * Cout;
    if (gid >= total) return;
    int oc = gid % Cout; int r = gid / Cout;
    int ox = r % Hout; r /= Hout;
    int oy = r % Hout; int n = r / Hout;

    float acc = bias[oc];
    for (int c = 0; c < Cin; ++c) {
#pragma unroll
        for (int ky = 0; ky < 3; ++ky) {
            int iy = 2 * oy - 1 + ky;
            if (iy < 0 || iy >= Hin) continue;
#pragma unroll
            for (int kx = 0; kx < 3; ++kx) {
                int ix = 2 * ox - 1 + kx;
                if (ix < 0 || ix >= Hin) continue;
                float xv = nchw
                    ? in[(long long)n * nStrideIn + ((size_t)c * Hin + iy) * Hin + ix]
                    : in[(long long)n * nStrideIn + ((size_t)iy * Hin + ix) * Cin + c];
                acc += xv * w[((oc * Cin + c) * 3 + ky) * 3 + kx];
            }
        }
    }
    out[((size_t)(n * Hout + oy) * Hout + ox) * Cout + oc] = acc;
}

// Row-wise LayerNorm over C; optional f32 and/or bf16 outputs.
__global__ void ln_rows_k(const float* __restrict__ in, float* __restrict__ outF,
                          unsigned short* __restrict__ outB,
                          const float* __restrict__ g, const float* __restrict__ b,
                          int P, int C)
{
    int r = blockIdx.x * blockDim.x + threadIdx.x;
    if (r >= P) return;
    const float* x = in + (size_t)r * C;
    float m = 0.f;
    for (int c = 0; c < C; ++c) m += x[c];
    m /= C;
    float v = 0.f;
    for (int c = 0; c < C; ++c) { float d = x[c] - m; v += d * d; }
    v /= C;
    float inv = rsqrtf(v + 1e-5f);
    for (int c = 0; c < C; ++c) {
        float o = (x[c] - m) * inv * g[c] + b[c];
        if (outF) outF[(size_t)r * C + c] = o;
        if (outB) outB[(size_t)r * C + c] = f2bf(o);
    }
}

// Row-wise L2 normalize -> bf16, same layout [P,C]
__global__ void l2n_bf_k(const float* __restrict__ in, unsigned short* __restrict__ out,
                         int P, int C)
{
    int r = blockIdx.x * blockDim.x + threadIdx.x;
    if (r >= P) return;
    const float* p = in + (size_t)r * C;
    float s = 0.f;
    for (int c = 0; c < C; ++c) s += p[c] * p[c];
    float inv = 1.0f / fmaxf(sqrtf(s), 1e-12f);
    for (int c = 0; c < C; ++c) out[(size_t)r * C + c] = f2bf(p[c] * inv);
}

// Row-wise L2 normalize -> bf16 transposed [C, P] (per-image blocks along P)
__global__ void l2n_T_bf_k(const float* __restrict__ in, unsigned short* __restrict__ out,
                           int P, int C)
{
    int r = blockIdx.x * blockDim.x + threadIdx.x;
    if (r >= P) return;
    const float* p = in + (size_t)r * C;
    float s = 0.f;
    for (int c = 0; c < C; ++c) s += p[c] * p[c];
    float inv = 1.0f / fmaxf(sqrtf(s), 1e-12f);
    for (int c = 0; c < C; ++c) out[(size_t)c * P + r] = f2bf(p[c] * inv);
}

// FsumT_bf[n][c2][c1] = f[n][c1][c2] + kv[n][c1][c2];  kv = f  (recurrence carry)
__global__ void fuse_kv_k(unsigned short* __restrict__ FsT, float* __restrict__ kv,
                          const float* __restrict__ f, int C, int n)
{
    int i = blockIdx.x * blockDim.x + threadIdx.x;
    if (i >= n) return;
    int CC = C * C;
    int img = i / CC, rem = i - img * CC;
    int c1 = rem / C, c2 = rem - c1 * C;
    float fv = f[i];
    float fs = fv + kv[i];
    kv[i] = fv;
    FsT[(size_t)img * CC + c2 * C + c1] = f2bf(fs);
}

__global__ void zero_k(float* __restrict__ p, int n)
{
    int i = blockIdx.x * blockDim.x + threadIdx.x;
    if (i < n) p[i] = 0.0f;
}

// f32 -> bf16 (same layout)
__global__ void cvt_bf16_k(const float* __restrict__ src, unsigned short* __restrict__ dst, int n)
{
    int i = blockIdx.x * blockDim.x + threadIdx.x;
    if (i < n) dst[i] = f2bf(src[i]);
}

// f32 [R,S] -> bf16 [S,R]
__global__ void cvtT_bf16_k(const float* __restrict__ src, unsigned short* __restrict__ dst,
                            int R, int S)
{
    int i = blockIdx.x * blockDim.x + threadIdx.x;
    if (i >= R * S) return;
    int r = i / S, c = i - r * S;
    dst[(size_t)c * R + r] = f2bf(src[i]);
}

// Head: out[n,t,j] = head_b[j] + sum_{c,hw} X[n,hw,c] * W[(c*256+hw)*4 + j]
__global__ void head_k(const float* __restrict__ X, const float* __restrict__ W,
                       const float* __restrict__ b, float* __restrict__ out, int t)
{
    __shared__ float red[256];
    int n = blockIdx.x >> 2;
    int j = blockIdx.x & 3;
    float s = 0.f;
    for (int d = threadIdx.x; d < 65536; d += 256) {
        int c = d >> 8, hw = d & 255;
        s += X[(size_t)n * 65536 + hw * 256 + c] * W[(size_t)d * 4 + j];
    }
    red[threadIdx.x] = s;
    __syncthreads();
    for (int off = 128; off > 0; off >>= 1) {
        if (threadIdx.x < off) red[threadIdx.x] += red[threadIdx.x + off];
        __syncthreads();
    }
    if (threadIdx.x == 0) out[n * 32 + t * 4 + j] = red[0] + b[j];
}

// ---------------------------------------------------------------------------
// Host side
// ---------------------------------------------------------------------------
struct SP {
    const float *conv_w, *conv_b, *ln_g, *ln_b;
    const float *q_w, *q_b, *q_g, *q_beta;
    const float *k_w, *k_b, *k_g, *k_beta;
    const float *v_w, *v_b, *v_g, *v_beta;
    const float *o_w, *o_b, *o_g, *o_beta;
    const float *fc1_w, *fc1_b, *fc2_w, *fc2_b;
};

extern "C" void kernel_launch(void* const* d_in, const int* in_sizes, int n_in,
                              void* d_out, int out_size, void* d_ws, size_t ws_size,
                              hipStream_t stream)
{
    (void)out_size; (void)ws_size;
    auto F = [&](int i) { return (const float*)d_in[i]; };
    const int X_ELEMS = 4 * 8 * 10 * 128 * 128; // 5,242,880 — unique size

    const float *x, *head_w, *head_b;
    SP st[3];
    if (n_in > 0 && in_sizes[0] == X_ELEMS) {
        // insertion-order flattening
        x = F(0);
        for (int s = 0; s < 3; ++s) {
            int b = 1 + 24 * s;
            st[s].conv_w = F(b + 0);  st[s].conv_b = F(b + 1);
            st[s].ln_g   = F(b + 2);  st[s].ln_b   = F(b + 3);
            st[s].q_w = F(b + 4); st[s].q_b = F(b + 5); st[s].q_g = F(b + 6); st[s].q_beta = F(b + 7);
            st[s].k_w = F(b + 8); st[s].k_b = F(b + 9); st[s].k_g = F(b + 10); st[s].k_beta = F(b + 11);
            st[s].v_w = F(b + 12); st[s].v_b = F(b + 13); st[s].v_g = F(b + 14); st[s].v_beta = F(b + 15);
            st[s].o_w = F(b + 16); st[s].o_b = F(b + 17); st[s].o_g = F(b + 18); st[s].o_beta = F(b + 19);
            st[s].fc1_w = F(b + 20); st[s].fc1_b = F(b + 21);
            st[s].fc2_w = F(b + 22); st[s].fc2_b = F(b + 23);
        }
        head_w = F(73); head_b = F(74);
    } else {
        // pytree alphabetical flattening
        head_b = F(0); head_w = F(1);
        for (int s = 0; s < 3; ++s) {
            int b = 2 + 24 * s;
            st[s].conv_b = F(b + 0);  st[s].conv_w = F(b + 1);
            st[s].fc1_b  = F(b + 2);  st[s].fc1_w  = F(b + 3);
            st[s].fc2_b  = F(b + 4);  st[s].fc2_w  = F(b + 5);
            st[s].k_b = F(b + 6); st[s].k_beta = F(b + 7); st[s].k_g = F(b + 8); st[s].k_w = F(b + 9);
            st[s].ln_b = F(b + 10); st[s].ln_g = F(b + 11);
            st[s].o_b = F(b + 12); st[s].o_beta = F(b + 13); st[s].o_g = F(b + 14); st[s].o_w = F(b + 15);
            st[s].q_b = F(b + 16); st[s].q_beta = F(b + 17); st[s].q_g = F(b + 18); st[s].q_w = F(b + 19);
            st[s].v_b = F(b + 20); st[s].v_beta = F(b + 21); st[s].v_g = F(b + 22); st[s].v_w = F(b + 23);
        }
        x = F(n_in - 1);
    }

    // ---- workspace bump allocator (units: float32 slots) ----
    float* w = (float*)d_ws;
    size_t o = 0;
    auto allocF = [&](size_t n) { float* p = w + o; o += n; return p; };
    auto allocH = [&](size_t nHalves) {
        unsigned short* p = (unsigned short*)(w + o);
        o += (nHalves + 1) / 2;
        return p;
    };

    float* XC   = allocF(1048576);             // conv->LN output (f32, residual)
    float* QF   = allocF(1048576);             // q f32 (pre-l2n); reused as xb f32
    float* KF   = allocF(1048576);             // k f32 (pre-l2n)
    float* FRAW = allocF(262144);              // f = k^T v (f32)
    float* SOUT = allocF(1048576);             // stage output (f32)
    float* KV0  = allocF(16384);
    float* KV1  = allocF(65536);
    float* KV2  = allocF(262144);
    float* kvp[3] = { KV0, KV1, KV2 };

    unsigned short* XCB = allocH(1048576);     // xc bf16 [P,C]
    unsigned short* QB  = allocH(1048576);     // q_norm bf16 [P,C]
    unsigned short* KTB = allocH(1048576);     // k_norm^T bf16 [C,P]
    unsigned short* VTB = allocH(1048576);     // v^T bf16 [C,P]
    unsigned short* FSB = allocH(262144);      // (f+kv)^T bf16 [4][C][C]
    unsigned short* YB  = allocH(1048576);     // y bf16 [P,C]
    unsigned short* XBB = allocH(1048576);     // xb(post-LN) bf16 [P,C]
    unsigned short* HB  = allocH(4194304);     // mlp hidden bf16 [P,4C]

    unsigned short *QW[3], *KW[3], *VW[3], *OW[3], *FC1T[3], *FC2T[3];
    static const int Cout_[3] = { 64, 128, 256 };
    for (int s = 0; s < 3; ++s) {
        int C = Cout_[s];
        QW[s]   = allocH((size_t)C * C);
        KW[s]   = allocH((size_t)C * C);
        VW[s]   = allocH((size_t)C * C);
        OW[s]   = allocH((size_t)C * C);
        FC1T[s] = allocH((size_t)4 * C * C);
        FC2T[s] = allocH((size_t)4 * C * C);
    }

    // ---- one-time per launch: zero kv carries, convert weights to bf16 ----
    {
        int nz = 16384 + 65536 + 262144;
        zero_k<<<(nz + 255) / 256, 256, 0, stream>>>(KV0, nz);
    }
    for (int s = 0; s < 3; ++s) {
        int C = Cout_[s], CC = C * C;
        cvt_bf16_k<<<(CC + 255) / 256, 256, 0, stream>>>(st[s].q_w, QW[s], CC);
        cvt_bf16_k<<<(CC + 255) / 256, 256, 0, stream>>>(st[s].k_w, KW[s], CC);
        cvt_bf16_k<<<(CC + 255) / 256, 256, 0, stream>>>(st[s].v_w, VW[s], CC);
        cvt_bf16_k<<<(CC + 255) / 256, 256, 0, stream>>>(st[s].o_w, OW[s], CC);
        cvtT_bf16_k<<<(4 * CC + 255) / 256, 256, 0, stream>>>(st[s].fc1_w, FC1T[s], C, 4 * C);
        cvtT_bf16_k<<<(4 * CC + 255) / 256, 256, 0, stream>>>(st[s].fc2_w, FC2T[s], 4 * C, C);
    }

    static const int Cin_[3] = { 10, 64, 128 };
    static const int Hin_[3] = { 128, 64, 32 };
    static const int Hout_[3] = { 64, 32, 16 };

    for (int t = 0; t < 8; ++t) {
        for (int s = 0; s < 3; ++s) {
            const int Ci = Cin_[s], C = Cout_[s], Hi = Hin_[s], Ho = Hout_[s];
            const int HWo = Ho * Ho;
            const int P = 4 * HWo;

            const float* cin_ptr;
            long long nStride;
            int nchw;
            if (s == 0) { cin_ptr = x + (size_t)t * 163840; nStride = 1310720; nchw = 1; }
            else        { cin_ptr = SOUT; nStride = (long long)Hi * Hi * Ci; nchw = 0; }

            int convTot = P * C;
            conv3x3_s2_k<<<(convTot + 255) / 256, 256, 0, stream>>>(
                cin_ptr, st[s].conv_w, st[s].conv_b, XC, Ci, Hi, C, Ho, nchw, nStride);
            // LN -> f32 (residual) + bf16 (GEMM A)
            ln_rows_k<<<(P + 255) / 256, 256, 0, stream>>>(XC, XC, XCB,
                st[s].ln_g, st[s].ln_b, P, C);

            dim3 gP(C / 64, P / 64, 1);
            // q,k: 1x1+BN -> f32 ; v: 1x1+BN+ReLU -> bf16 transposed [C,P]
            gemm_bf16_wmma<1, 0><<<gP, 128, 0, stream>>>(XCB, QW[s], QF, P, C, C,
                C, C, C, 0, 0, 0, st[s].q_b, st[s].q_g, st[s].q_beta, nullptr, 0.f);
            gemm_bf16_wmma<1, 0><<<gP, 128, 0, stream>>>(XCB, KW[s], KF, P, C, C,
                C, C, C, 0, 0, 0, st[s].k_b, st[s].k_g, st[s].k_beta, nullptr, 0.f);
            gemm_bf16_wmma<2, 2><<<gP, 128, 0, stream>>>(XCB, VW[s], VTB, P, C, C,
                C, C, P, 0, 0, 0, st[s].v_b, st[s].v_g, st[s].v_beta, nullptr, 0.f);

            l2n_bf_k<<<(P + 255) / 256, 256, 0, stream>>>(QF, QB, P, C);
            l2n_T_bf_k<<<(P + 255) / 256, 256, 0, stream>>>(KF, KTB, P, C);

            // f = k_norm^T v  (per image): A=KTB [C,P], B=VTB [C,P], K=HWo
            dim3 gF(C / 64, C / 64, 4);
            gemm_bf16_wmma<0, 0><<<gF, 128, 0, stream>>>(KTB, VTB, FRAW, C, C, HWo,
                P, P, C, HWo, HWo, (long long)C * C,
                nullptr, nullptr, nullptr, nullptr, 0.f);

            int fcnt = 4 * C * C;
            fuse_kv_k<<<(fcnt + 255) / 256, 256, 0, stream>>>(FSB, kvp[s], FRAW, C, fcnt);

            // y = q_norm (f+kv): A=QB [P,C] per image, B=FsumT [C,C] -> YB bf16
            dim3 gY(C / 64, HWo / 64, 4);
            gemm_bf16_wmma<0, 1><<<gY, 128, 0, stream>>>(QB, FSB, YB, HWo, C, C,
                C, C, C, (long long)HWo * C, (long long)C * C, (long long)HWo * C,
                nullptr, nullptr, nullptr, nullptr, 0.f);

            // o-proj: relu(bn(y Wo^T + b)) + 2*xc -> f32 (reuse QF as xb)
            gemm_bf16_wmma<3, 0><<<gP, 128, 0, stream>>>(YB, OW[s], QF, P, C, C,
                C, C, C, 0, 0, 0, st[s].o_b, st[s].o_g, st[s].o_beta, XC, 2.0f);

            // LN -> bf16 only (MLP input)
            ln_rows_k<<<(P + 255) / 256, 256, 0, stream>>>(QF, nullptr, XBB,
                st[s].ln_g, st[s].ln_b, P, C);

            // MLP: fc1+GELU -> bf16 hidden; fc2+bias -> f32 stage output
            dim3 gF1((4 * C) / 64, P / 64, 1);
            gemm_bf16_wmma<4, 1><<<gF1, 128, 0, stream>>>(XBB, FC1T[s], HB, P, 4 * C, C,
                C, C, 4 * C, 0, 0, 0, st[s].fc1_b, nullptr, nullptr, nullptr, 0.f);
            dim3 gF2(C / 64, P / 64, 1);
            gemm_bf16_wmma<5, 0><<<gF2, 128, 0, stream>>>(HB, FC2T[s], SOUT, P, C, 4 * C,
                4 * C, 4 * C, C, 0, 0, 0, st[s].fc2_b, nullptr, nullptr, nullptr, 0.f);
        }
        head_k<<<16, 256, 0, stream>>>(SOUT, head_w, head_b, (float*)d_out, t);
    }
}
